// Policy_11227044512416
// MI455X (gfx1250) — matrix operand order repaired and checked by reading, not compiled, not run on hardware
//
#include <hip/hip_runtime.h>
#include <cfloat>
#include <cmath>
#include <cstdint>

typedef __attribute__((ext_vector_type(16))) _Float16 v16h;
typedef __attribute__((ext_vector_type(8)))  _Float16 v8h;
typedef __attribute__((ext_vector_type(8)))  float    v8f;

// ---------------------------------------------------------------------------
// WMMA GEMM:  C[M,Nout] (+)= A[M,K] * W[K,Nout]   (f32 memory, f16 WMMA MACs)
//
// Requirements (guaranteed by the host code):
//   - A has lda >= Kpad columns readable per row, rows 16B-aligned (lda%8==0)
//   - Kpad = round_up(K,32), power-of-two (32 or 64)
//   - M % 16 == 0, Nout % 16 == 0
// Block = 128 threads = 4 waves: 4 consecutive m-tiles x one 16-wide n-tile.
// Weight tile staged to LDS (zero-padded, transposed [n][k], stride 72).
// Inner loop: 4x global b128 (A) + 2x ds b128 (B) + 1 WMMA, no predication.
// ---------------------------------------------------------------------------
#define BSTRIDE 72

__global__ void wmma_gemm_k(const float* __restrict__ A, int lda, int K, int Kpad,
                            const float* __restrict__ W, int ldw,
                            float* __restrict__ C, int ldc,
                            int M, int Nout, int accumulate)
{
    __shared__ _Float16 Bs[16 * BSTRIDE];   // [n][k], Kpad <= 64

    const int lane  = threadIdx.x & 31;
    const int wave  = threadIdx.x >> 5;
    const int tilesN = Nout >> 4;
    const int n0   = (blockIdx.x % tilesN) << 4;
    const int bidM = blockIdx.x / tilesN;

    // Cooperative stage: W[k, n0+n] -> Bs[n*BSTRIDE + k], f16, zero pad k>=K
    for (int idx = threadIdx.x; idx < Kpad * 16; idx += blockDim.x) {
        int k = idx & (Kpad - 1);           // Kpad is a power of two
        int n = idx / Kpad;
        float v = (k < K) ? W[(size_t)k * ldw + n0 + n] : 0.0f;
        Bs[n * BSTRIDE + k] = (_Float16)v;
    }
    __syncthreads();

    const int m0 = (bidM * 4 + wave) << 4;
    if (m0 >= M) return;                    // wave-uniform: EXEC stays full
    const int lm = lane & 15;
    const int hi = lane >> 4;

    v8f acc = {};
    if (accumulate) {
#pragma unroll
        for (int i = 0; i < 8; ++i)
            acc[i] = C[(size_t)(m0 + i + 8 * hi) * ldc + n0 + lm];
    }

    const float* Arow = A + (size_t)(m0 + lm) * lda;

    for (int k0 = 0; k0 < Kpad; k0 += 32) {
        if (k0 + 32 < Kpad)
            __builtin_prefetch(Arow + k0 + 32, 0, 0);
        // A fragment: rows of the 16-bit 16x32 layout = two 8-float runs
        const float4* r1 = (const float4*)(Arow + k0 + 8 * hi);       // k0+8hi..+7
        const float4* r2 = (const float4*)(Arow + k0 + 16 + 8 * hi);  // k0+16+8hi..+7
        float4 a0 = r1[0], a1 = r1[1];
        float4 a2 = r2[0], a3 = r2[1];
        v16h a;
        a[0]  = (_Float16)a0.x; a[1]  = (_Float16)a0.y;
        a[2]  = (_Float16)a0.z; a[3]  = (_Float16)a0.w;
        a[4]  = (_Float16)a1.x; a[5]  = (_Float16)a1.y;
        a[6]  = (_Float16)a1.z; a[7]  = (_Float16)a1.w;
        a[8]  = (_Float16)a2.x; a[9]  = (_Float16)a2.y;
        a[10] = (_Float16)a2.z; a[11] = (_Float16)a2.w;
        a[12] = (_Float16)a3.x; a[13] = (_Float16)a3.y;
        a[14] = (_Float16)a3.z; a[15] = (_Float16)a3.w;
        // B fragment: two 8-f16 runs for column lm, k = k0+8*hi..+7 and k0+16+8*hi..+7
        const _Float16* bp = &Bs[lm * BSTRIDE + k0 + 8 * hi];
        v8h b0 = *(const v8h*)(bp);
        v8h b1 = *(const v8h*)(bp + 16);
        v16h b;
#pragma unroll
        for (int e = 0; e < 8; ++e) { b[e] = b0[e]; b[e + 8] = b1[e]; }

        acc = __builtin_amdgcn_wmma_f32_16x16x32_f16(
                  false, a, false, b, (short)0, acc, false, false);
    }

#pragma unroll
    for (int i = 0; i < 8; ++i)
        C[(size_t)(m0 + i + 8 * hi) * ldc + n0 + lm] = acc[i];
}

// ---------------------------------------------------------------------------
// Elementwise helpers
// ---------------------------------------------------------------------------
__global__ void fill_f32(float* p, float v, size_t n)
{
    size_t t = (size_t)blockIdx.x * blockDim.x + threadIdx.x;
    if (t < n) p[t] = v;
}

// XP[n, 0..31] = {x[n, 0..6], 0...}
__global__ void pad_x_k(const float* __restrict__ x, float* __restrict__ XP, int N)
{
    size_t t = (size_t)blockIdx.x * blockDim.x + threadIdx.x;
    if (t >= (size_t)N * 32) return;
    int n = (int)(t >> 5), c = (int)(t & 31);
    XP[t] = (c < 7) ? x[(size_t)n * 7 + c] : 0.0f;
}

__global__ void bias_leaky_k(float* X, const float* __restrict__ b, int N, int C,
                             float slope)
{
    size_t t = (size_t)blockIdx.x * blockDim.x + threadIdx.x;
    if (t >= (size_t)N * C) return;
    int c = (int)(t % C);
    float v = X[t] + b[c];
    X[t] = v > 0.0f ? v : slope * v;
}

// ---------------------------------------------------------------------------
// GAT pieces
// ---------------------------------------------------------------------------
__device__ inline float atomicMaxFloat(float* addr, float value)
{
    return (value >= 0.0f)
        ? __int_as_float(atomicMax((int*)addr, __float_as_int(value)))
        : __uint_as_float(atomicMin((unsigned int*)addr, __float_as_uint(value)));
}

// a_src[n,h] = sum_c h[n,h,c]*att_src[h,c];  a_dst likewise
__global__ void gat_node_attn(const float* __restrict__ H,
                              const float* __restrict__ attS,
                              const float* __restrict__ attD,
                              float* __restrict__ AS, float* __restrict__ AD,
                              int N, int cout)
{
    int t = blockIdx.x * blockDim.x + threadIdx.x;
    if (t >= 2 * N) return;
    int n = t >> 1, h = t & 1;
    const float* hp = H + (size_t)n * 2 * cout + (size_t)h * cout;
    float s = 0.f, d = 0.f;
    for (int c = 0; c < cout; ++c) {
        float v = hp[c];
        s += v * attS[h * cout + c];
        d += v * attD[h * cout + c];
    }
    AS[t] = s;
    AD[t] = d;
}

// segment_max of leaky(a_src[s]+a_dst[d], 0.2) over dst
__global__ void gat_edge_max(const int* __restrict__ src, const int* __restrict__ dst,
                             const float* __restrict__ AS, const float* __restrict__ AD,
                             float* __restrict__ MX, int E)
{
    int e = blockIdx.x * blockDim.x + threadIdx.x;
    if (e >= E) return;
    int s = src[e], d = dst[e];
#pragma unroll
    for (int h = 0; h < 2; ++h) {
        float v = AS[s * 2 + h] + AD[d * 2 + h];
        v = v > 0.0f ? v : 0.2f * v;
        atomicMaxFloat(&MX[d * 2 + h], v);
    }
}

// One WAVE per edge: lanes cover the 2*cout channels -> coalesced gather of
// h[src,:] and coalesced f32 atomics into acc[dst,:].
__global__ void gat_edge_acc(const int* __restrict__ src, const int* __restrict__ dst,
                             const float* __restrict__ AS, const float* __restrict__ AD,
                             const float* __restrict__ MX, float* __restrict__ DN,
                             const float* __restrict__ H, float* __restrict__ AC,
                             int E, int cout)
{
    int t = blockIdx.x * blockDim.x + threadIdx.x;
    int e = t >> 5;
    int lane = t & 31;
    if (e >= E) return;
    int s = src[e], d = dst[e];
    float v0 = AS[s * 2 + 0] + AD[d * 2 + 0];
    v0 = v0 > 0.0f ? v0 : 0.2f * v0;
    float ex0 = __expf(v0 - MX[d * 2 + 0]);
    float v1 = AS[s * 2 + 1] + AD[d * 2 + 1];
    v1 = v1 > 0.0f ? v1 : 0.2f * v1;
    float ex1 = __expf(v1 - MX[d * 2 + 1]);
    if (lane < 2) atomicAdd(&DN[d * 2 + lane], lane ? ex1 : ex0);
    const float* hs = H  + (size_t)s * 2 * cout;
    float*       ap = AC + (size_t)d * 2 * cout;
    for (int c = lane; c < 2 * cout; c += 32)
        atomicAdd(&ap[c], (c >= cout ? ex1 : ex0) * hs[c]);
}

// out[n,c] = 0.5*(acc[n,0,c]/den[n,0] + acc[n,1,c]/den[n,1]) + b[c]
__global__ void gat_finalize(const float* __restrict__ AC, const float* __restrict__ DN,
                             const float* __restrict__ b, float* __restrict__ out,
                             int N, int cout)
{
    size_t t = (size_t)blockIdx.x * blockDim.x + threadIdx.x;
    if (t >= (size_t)N * cout) return;
    int n = (int)(t / cout), c = (int)(t % cout);
    float d0 = DN[n * 2 + 0], d1 = DN[n * 2 + 1];
    float v0 = d0 > 0.0f ? AC[(size_t)n * 2 * cout + c]        / d0 : 0.0f;
    float v1 = d1 > 0.0f ? AC[(size_t)n * 2 * cout + cout + c] / d1 : 0.0f;
    out[(size_t)n * cout + c] = 0.5f * (v0 + v1) + b[c];
}

// pooled[g,c] += x4[n,c]   (attn softmax over size-1 dim == 1.0)
__global__ void pool_k(const float* __restrict__ X4, const int* __restrict__ batch,
                       float* __restrict__ P, int N)
{
    size_t t = (size_t)blockIdx.x * blockDim.x + threadIdx.x;
    if (t >= (size_t)N * 64) return;
    int n = (int)(t >> 6), c = (int)(t & 63);
    atomicAdd(&P[batch[n] * 64 + c], X4[(size_t)n * 64 + c]);
}

// Tiny MLP head: one thread per graph (64 graphs)
__global__ void head_k(const float* __restrict__ P,
                       const float* __restrict__ W1, const float* __restrict__ b1,
                       const float* __restrict__ W2, const float* __restrict__ b2,
                       const float* __restrict__ Wm, const float* __restrict__ bm,
                       const float* __restrict__ Ws, const float* __restrict__ bs,
                       float* __restrict__ out)
{
    int g = threadIdx.x;
    if (g >= 64) return;
    const float* pg = P + g * 64;
    float h1[32];
    for (int j = 0; j < 32; ++j) {
        float s = b1[j];
        for (int i = 0; i < 64; ++i) s += pg[i] * W1[i * 32 + j];
        h1[j] = s > 0.0f ? s : 0.01f * s;
    }
    float h2[8];
    for (int j = 0; j < 8; ++j) {
        float s = b2[j];
        for (int i = 0; i < 32; ++i) s += h1[i] * W2[i * 8 + j];
        h2[j] = s > 0.0f ? s : 0.01f * s;
    }
    float mu = bm[0];
    for (int i = 0; i < 8; ++i) mu += h2[i] * Wm[i];
    mu = tanhf(mu);
    float sg = bs[0];
    for (int i = 0; i < 8; ++i) sg += h2[i] * Ws[i];
    float sp = sg > 20.0f ? sg : log1pf(__expf(sg));  // softplus
    out[g]      = mu;
    out[64 + g] = sp;
}

// ---------------------------------------------------------------------------
// Host orchestration
// ---------------------------------------------------------------------------
extern "C" void kernel_launch(void* const* d_in, const int* in_sizes, int n_in,
                              void* d_out, int out_size, void* d_ws, size_t ws_size,
                              hipStream_t stream)
{
    const float* x     = (const float*)d_in[0];
    const int*   ei    = (const int*)d_in[1];
    const int*   batch = (const int*)d_in[2];
    const int N = in_sizes[2];            // 50000 (divisible by 16)
    const int E = in_sizes[1] / 2;        // 1600000
    const int* src = ei;
    const int* dst = ei + E;

    auto fp = [&](int i) { return (const float*)d_in[i]; };

    // ---- workspace carve ----
    char* base = (char*)d_ws;
    size_t off = 0;
    auto alloc = [&](size_t bytes) -> float* {
        off = (off + 255) & ~(size_t)255;
        float* p = (float*)(base + off);
        off += bytes;
        return p;
    };
    float* HF  = alloc((size_t)N * 128 * 4);
    float* HB  = alloc((size_t)N * 128 * 4);
    float* ACF = alloc((size_t)N * 128 * 4);
    float* ACB = alloc((size_t)N * 128 * 4);
    float* XF  = alloc((size_t)N * 64 * 4);
    float* XB  = alloc((size_t)N * 64 * 4);
    float* XP  = alloc((size_t)N * 32 * 4);   // x zero-padded to 32 cols
    float* X1  = alloc((size_t)N * 32 * 4);
    float* X2  = alloc((size_t)N * 32 * 4);
    float* X3  = alloc((size_t)N * 32 * 4);
    float* X4  = alloc((size_t)N * 64 * 4);
    float* ASF = alloc((size_t)N * 2 * 4);
    float* ADF = alloc((size_t)N * 2 * 4);
    float* ASB = alloc((size_t)N * 2 * 4);
    float* ADB = alloc((size_t)N * 2 * 4);
    float* MXF = alloc((size_t)N * 2 * 4);
    float* DNF = alloc((size_t)N * 2 * 4);
    float* MXB = alloc((size_t)N * 2 * 4);
    float* DNB = alloc((size_t)N * 2 * 4);
    float* POOL = alloc(64 * 64 * 4);

    // C (+)= A[M,K] * W, with A having lda (>= Kpad) readable columns
    auto gemm = [&](const float* A, int lda, int K, const float* W, int ldw,
                    float* C, int ldc, int Nout, int acc) {
        int Kpad = (K + 31) & ~31;                 // 32 or 64
        int tilesN = Nout / 16;
        int blocksM = (N / 16 + 3) / 4;            // 4 m-tiles per block
        wmma_gemm_k<<<blocksM * tilesN, 128, 0, stream>>>(
            A, lda, K, Kpad, W, ldw, C, ldc, N, Nout, acc);
    };

    auto run_gat = [&](const float* Xin, int lda, int cin, int pidx,
                       const int* s, const int* d,
                       float* H, float* AC, float* AS, float* AD,
                       float* MX, float* DN, float* Xout, int cout) {
        const float* Wg   = fp(pidx + 0);
        const float* attS = fp(pidx + 1);
        const float* attD = fp(pidx + 2);
        const float* bg   = fp(pidx + 3);
        const int twoC = 2 * cout;
        gemm(Xin, lda, cin, Wg, twoC, H, twoC, twoC, 0);
        gat_node_attn<<<(2 * N + 255) / 256, 256, 0, stream>>>(H, attS, attD, AS, AD, N, cout);
        fill_f32<<<(2 * N + 255) / 256, 256, 0, stream>>>(MX, -FLT_MAX, (size_t)2 * N);
        fill_f32<<<(2 * N + 255) / 256, 256, 0, stream>>>(DN, 0.0f, (size_t)2 * N);
        fill_f32<<<(int)(((size_t)N * twoC + 255) / 256), 256, 0, stream>>>(AC, 0.0f, (size_t)N * twoC);
        gat_edge_max<<<(E + 255) / 256, 256, 0, stream>>>(s, d, AS, AD, MX, E);
        gat_edge_acc<<<(int)(((size_t)E * 32 + 255) / 256), 256, 0, stream>>>(
            s, d, AS, AD, MX, DN, H, AC, E, cout);
        gat_finalize<<<(int)(((size_t)N * cout + 255) / 256), 256, 0, stream>>>(AC, DN, bg, Xout, N, cout);
    };

    // ---- pad x to 32 columns (makes every GEMM A-operand tail-free) ----
    pad_x_k<<<(int)(((size_t)N * 32 + 255) / 256), 256, 0, stream>>>(x, XP, N);

    // ---- layers 1..3: bi-GAT(32) + agg(64->32) ----
    const int fwdIdx[4] = {3, 11, 19, 27};
    const int bwdIdx[4] = {7, 15, 23, 31};
    const int aggIdx[4] = {35, 37, 39, 41};
    float* Xout13[3] = {X1, X2, X3};
    const float* Xin = XP;
    int lin = 32, cin = 7;
    for (int l = 0; l < 3; ++l) {
        const int cout = 32;
        run_gat(Xin, lin, cin, fwdIdx[l], src, dst, HF, ACF, ASF, ADF, MXF, DNF, XF, cout);
        run_gat(Xin, lin, cin, bwdIdx[l], dst, src, HB, ACB, ASB, ADB, MXB, DNB, XB, cout);
        const float* Wa = fp(aggIdx[l]);
        const float* ba = fp(aggIdx[l] + 1);
        float* Xn = Xout13[l];
        gemm(XF, cout, cout, Wa,           32, Xn, 32, 32, 0);
        gemm(XB, cout, cout, Wa + 32 * 32, 32, Xn, 32, 32, 1);
        bias_leaky_k<<<(int)(((size_t)N * 32 + 255) / 256), 256, 0, stream>>>(Xn, ba, N, 32, 0.01f);
        Xin = Xn;
        lin = 32;
        cin = 32;
    }

    // ---- layer 4: bi-GAT(64) + agg over cat(xf, xb, x, x1, x2, x3) ----
    {
        const int cout = 64;
        run_gat(X3, 32, 32, fwdIdx[3], src, dst, HF, ACF, ASF, ADF, MXF, DNF, XF, cout);
        run_gat(X3, 32, 32, bwdIdx[3], dst, src, HB, ACB, ASB, ADB, MXB, DNB, XB, cout);
        const float* Wa = fp(aggIdx[3]);      // [231, 64] row-major
        const float* ba = fp(aggIdx[3] + 1);
        gemm(XF, 64, 64, Wa +   0 * 64, 64, X4, 64, 64, 0);
        gemm(XB, 64, 64, Wa +  64 * 64, 64, X4, 64, 64, 1);
        gemm(XP, 32,  7, Wa + 128 * 64, 64, X4, 64, 64, 1);
        gemm(X1, 32, 32, Wa + 135 * 64, 64, X4, 64, 64, 1);
        gemm(X2, 32, 32, Wa + 167 * 64, 64, X4, 64, 64, 1);
        gemm(X3, 32, 32, Wa + 199 * 64, 64, X4, 64, 64, 1);
        bias_leaky_k<<<(int)(((size_t)N * 64 + 255) / 256), 256, 0, stream>>>(X4, ba, N, 64, 0.01f);
    }

    // ---- readout + head ----
    fill_f32<<<(64 * 64 + 255) / 256, 256, 0, stream>>>(POOL, 0.0f, 64 * 64);
    pool_k<<<(int)(((size_t)N * 64 + 255) / 256), 256, 0, stream>>>(X4, batch, POOL, N);
    head_k<<<1, 64, 0, stream>>>(POOL,
                                 fp(45), fp(46),   // mlp1
                                 fp(47), fp(48),   // mlp2
                                 fp(49), fp(50),   // mu
                                 fp(51), fp(52),   // sigma
                                 (float*)d_out);
}
